// TemporalAttention_30099130810846
// MI455X (gfx1250) — compile-verified, matrix operand order checked by reading, not run on hardware
//
#include <hip/hip_runtime.h>
#include <hip/hip_bf16.h>

// MI455X / gfx1250 (wave32) implementation.
// Pass 1: bf16 WMMA GEMM (256 -> 64) + ReLU + w2-dot fused -> per-(slot,pixel) scores.
// Pass 2: softmax over 8 slots (once per pixel, via LDS) + fp32 weighted sum (HBM-bound).

typedef __attribute__((ext_vector_type(16))) __bf16 v16bf;
typedef __attribute__((ext_vector_type(8)))  __bf16 v8bf;
typedef __attribute__((ext_vector_type(8)))  float  v8f;

#define HW_    4096   // H*W
#define C_     256
#define CH_    64
#define NSLOT  8      // 2T
#define TQ_    4      // T

// ---------------------------------------------------------------------------
// Kernel 1: scores[slot][p] for slot = b*8+t2 (t2<4 -> spike, else mem)
// Block = 128 threads (4 waves). Each wave: one 16-pixel tile of one slot.
// Grid = 64 slots * 64 blocks/slot = 4096 blocks (each block covers 64 pixels).
// ---------------------------------------------------------------------------
__global__ __launch_bounds__(128) void ta_scores_kernel(
    const float* __restrict__ spike, const float* __restrict__ mem,
    const float* __restrict__ w1, const float* __restrict__ b1,
    const float* __restrict__ w2, const float* __restrict__ b2,
    float* __restrict__ scores)
{
    __shared__ __attribute__((aligned(16))) __bf16 lw1[CH_ * C_];  // 32 KB, [d][c]
    __shared__ float lb1[CH_];
    __shared__ float lw2[CH_];

    const int tid = threadIdx.x;

    // Stage W1 (fp32 -> bf16) into LDS; tiny, hot in L2 across blocks.
    for (int i = tid; i < CH_ * C_; i += 128) lw1[i] = (__bf16)w1[i];
    if (tid < CH_)            lb1[tid]        = b1[tid];
    else if (tid < 2 * CH_)   lw2[tid - CH_]  = w2[tid - CH_];
    __syncthreads();

    const int slot  = blockIdx.x >> 6;         // 0..63 = b*8 + t2
    const int pblk  = (blockIdx.x & 63) << 6;  // 64-pixel block base
    const int wave  = tid >> 5;
    const int lane  = tid & 31;
    const int lp    = lane & 15;               // pixel within tile / row within A
    const int khalf = lane >> 4;               // 0: K 0-7,16-23   1: K 8-15,24-31
    const int p     = pblk + (wave << 4) + lp; // this lane's pixel (B-matrix col)

    const int b  = slot >> 3;
    const int t2 = slot & 7;
    const float* __restrict__ sb =
        (t2 < TQ_ ? spike : mem) + ((b * TQ_ + (t2 & 3)) * C_) * HW_;

    v8f acc[4] = {};   // hidden[64 x 16] : 4 M-tiles of 16 rows

    #pragma unroll
    for (int ks = 0; ks < 8; ++ks) {           // K = 256 in steps of 32
        const int k0 = ks * 32;
        const int co = k0 + (khalf << 3);

        // ---- B operand (32x16 bf16): column = pixel, per-lane k-strided rows.
        // Loads are fully coalesced across the 16 pixel lanes (64B / row).
        v16bf bm;
        #pragma unroll
        for (int j = 0; j < 8; ++j) {
            bm[j]     = (__bf16)sb[(co + j)      * HW_ + p];
            bm[8 + j] = (__bf16)sb[(co + 16 + j) * HW_ + p];
        }
        // Token prefetch of next K-step's rows (global_prefetch_b8).
        if (ks < 7) __builtin_prefetch(&sb[(co + 32) * HW_ + p], 0, 3);

        // ---- 4 M-tiles of A (16x32 bf16) from LDS, one WMMA each.
        #pragma unroll
        for (int mt = 0; mt < 4; ++mt) {
            const int d = (mt << 4) + lp;      // A row
            const v8bf* av = reinterpret_cast<const v8bf*>(&lw1[(d << 8) + co]);
            const v8bf lo = av[0];             // K = co .. co+7     (ds_load_b128)
            const v8bf hi = av[2];             // K = co+16 .. co+23 (ds_load_b128)
            v16bf am;
            #pragma unroll
            for (int e = 0; e < 8; ++e) { am[e] = lo[e]; am[8 + e] = hi[e]; }

            acc[mt] = __builtin_amdgcn_wmma_f32_16x16x32_bf16(
                false, am, false, bm, (short)0, acc[mt], false, false);
        }
    }

    // ---- Epilogue: relu(hidden + b1) . w2, per pixel.
    // C/D layout: lane 0-15 col=pixel, VGPR r -> row M=r; lanes 16-31 -> M=r+8.
    const int dbase = khalf << 3;
    float partial = 0.f;
    #pragma unroll
    for (int mt = 0; mt < 4; ++mt) {
        #pragma unroll
        for (int r = 0; r < 8; ++r) {
            const int d = (mt << 4) + r + dbase;
            float h = acc[mt][r] + lb1[d];
            h = h > 0.f ? h : 0.f;
            partial += h * lw2[d];
        }
    }
    // Combine the two half-rows (lane l and l+16 hold the same pixel).
    partial += __shfl_xor(partial, 16, 32);

    if (lane < 16) scores[slot * HW_ + p] = partial + b2[0];
}

// ---------------------------------------------------------------------------
// Kernel 2: softmax over 8 slots + weighted channel sum. Pure HBM streaming.
// Block = 256 threads, covers 32 pixels x 256 channels. Grid = 8 * 128.
// ---------------------------------------------------------------------------
__global__ __launch_bounds__(256) void ta_combine_kernel(
    const float* __restrict__ spike, const float* __restrict__ mem,
    const float* __restrict__ scores, float* __restrict__ out)
{
    __shared__ float wgt[NSLOT][32];

    const int tid = threadIdx.x;
    const int b   = blockIdx.x >> 7;           // 128 blocks per batch
    const int pb  = (blockIdx.x & 127) << 5;   // 32-pixel base within HW

    // Softmax once per pixel (first wave only), result into LDS.
    if (tid < 32) {
        const int p = pb + tid;
        float s[NSLOT];
        float m = -3.4e38f;
        #pragma unroll
        for (int j = 0; j < NSLOT; ++j) {
            s[j] = scores[(b * NSLOT + j) * HW_ + p];
            m = fmaxf(m, s[j]);
        }
        float sum = 0.f;
        #pragma unroll
        for (int j = 0; j < NSLOT; ++j) { s[j] = __expf(s[j] - m); sum += s[j]; }
        const float inv = 1.f / sum;
        #pragma unroll
        for (int j = 0; j < NSLOT; ++j) wgt[j][tid] = s[j] * inv;
    }
    __syncthreads();

    const int lane = tid & 31;   // pixel lane (coalesced 128B transactions)
    const int crow = tid >> 5;   // 0..7 -> channels crow, crow+8, ...
    const int p    = pb + lane;

    float w[NSLOT];
    #pragma unroll
    for (int j = 0; j < NSLOT; ++j) w[j] = wgt[j][lane];

    #pragma unroll 4
    for (int c = crow; c < C_; c += 8) {
        float acc = 0.f;
        #pragma unroll
        for (int j = 0; j < NSLOT; ++j) {
            const float* __restrict__ src = (j < TQ_) ? spike : mem;
            acc = fmaf(src[(((b * TQ_ + (j & 3)) * C_) + c) * HW_ + p], w[j], acc);
        }
        out[(b * C_ + c) * HW_ + p] = acc;
    }
}

extern "C" void kernel_launch(void* const* d_in, const int* in_sizes, int n_in,
                              void* d_out, int out_size, void* d_ws, size_t ws_size,
                              hipStream_t stream) {
    const float* spike = (const float*)d_in[0];   // [8,4,256,64,64]
    const float* mem   = (const float*)d_in[1];   // [8,4,256,64,64]
    const float* w1    = (const float*)d_in[2];   // [64,256]
    const float* b1    = (const float*)d_in[3];   // [64]
    const float* w2    = (const float*)d_in[4];   // [1,64]
    const float* b2    = (const float*)d_in[5];   // [1]
    float*       out   = (float*)d_out;           // [8,256,64,64]
    float*       scores = (float*)d_ws;           // 64 * 4096 floats = 1 MB scratch

    // Pass 1: 64 slots * 64 blocks (64 px each), 4 waves/block.
    ta_scores_kernel<<<dim3(4096), dim3(128), 0, stream>>>(
        spike, mem, w1, b1, w2, b2, scores);

    // Pass 2: 8 batches * 128 strips of 32 px, 256 threads/block.
    ta_combine_kernel<<<dim3(1024), dim3(256), 0, stream>>>(
        spike, mem, scores, out);
}
